// SlotAttentionLayer_66468913873544
// MI455X (gfx1250) — matrix operand order repaired and checked by reading, not compiled, not run on hardware
//
#include <hip/hip_runtime.h>
#include <hip/hip_bf16.h>
#include <math.h>

#define BN 64
#define SN 16
#define FN 4096
#define DIN 512
#define DSL 512
#define DKV 512
#define DHID 2048
#define LN_EPS_F 1e-5f
#define EPS_A 1e-8f

typedef __attribute__((ext_vector_type(16))) __bf16 v16bf;
typedef __attribute__((ext_vector_type(8)))  float  v8f;

union FragAB { v16bf v; uint4 q[2]; };

static __device__ __forceinline__ unsigned short f2bf(float x) {
    union { float f; unsigned int u; } c; c.f = x;
    unsigned int r = (c.u + 0x7FFFu + ((c.u >> 16) & 1u)) >> 16;
    return (unsigned short)r;
}

static __device__ __forceinline__ float wred(float v) {
#pragma unroll
    for (int o = 16; o; o >>= 1) v += __shfl_xor(v, o, 32);
    return v;
}

// ---------------- per-row LayerNorm stats for features (mu, rstd) ----------
__global__ void k_feat_stats(const float* __restrict__ x, float* __restrict__ stats) {
    int w = threadIdx.x >> 5, l = threadIdx.x & 31;
    long long row = (long long)blockIdx.x * 8 + w;
    const float* xr = x + row * DIN;
    float s = 0.f, sq = 0.f;
    for (int i = l; i < DIN; i += 32) { float v = xr[i]; s += v; sq += v * v; }
    s = wred(s); sq = wred(sq);
    if (l == 0) {
        float mu = s / (float)DIN;
        float var = sq / (float)DIN - mu * mu;
        stats[row * 2]     = mu;
        stats[row * 2 + 1] = rsqrtf(var + LN_EPS_F);
    }
}

// ---------------- convert w_k / w_v to bf16 --------------------------------
__global__ void k_cvt2(const float* __restrict__ a, const float* __restrict__ b,
                       unsigned short* __restrict__ oa, unsigned short* __restrict__ ob) {
    int i = blockIdx.x * 256 + threadIdx.x;
    if (i < DKV * DIN) { oa[i] = f2bf(a[i]); ob[i] = f2bf(b[i]); }
}

// ---------------- fused LN + K/V projection, bf16 WMMA, 2x2 reg blocking ---
// Macro tile 64 rows x 128 cols per 256-thread block (8 waves, 2x4 wave grid,
// each wave owns a 2x2 grid of 16x16 WMMA tiles).
// C cols [0,512) -> keys (row-major [f,d]); [512,1024) -> values transposed [d,f].
__global__ void k_kv(const float* __restrict__ feat, const float* __restrict__ stats,
                     const float* __restrict__ g, const float* __restrict__ bia,
                     const unsigned short* __restrict__ wk, const unsigned short* __restrict__ wv,
                     unsigned short* __restrict__ kbf, unsigned short* __restrict__ vtbf) {
    __shared__ unsigned short As[64 * 32];   // 4 KB
    __shared__ unsigned short Bs[128 * 32];  // 8 KB
    int t = threadIdx.x;
    int w = t >> 5, l = t & 31;
    long long R0 = (long long)blockIdx.x * 64;
    int C0 = blockIdx.y * 128;

    // A staging: 64 rows x 32 k fp32 = 512 float4; 2 float4 per thread
    int ar0 = t >> 3,  ak0 = (t & 7) * 4;
    int ar1 = ar0 + 32;
    float mu0 = stats[(R0 + ar0) * 2], rs0 = stats[(R0 + ar0) * 2 + 1];
    float mu1 = stats[(R0 + ar1) * 2], rs1 = stats[(R0 + ar1) * 2 + 1];
    // B staging: 128 rows x 32 k bf16 = 512 uint4; 2 uint4 per thread
    int br0 = t >> 2, bk0 = (t & 3) * 8;
    int br1 = br0 + 64;
    int bc0 = C0 + br0, bc1 = C0 + br1;
    const unsigned short* wsrc0 = (bc0 < DKV) ? (wk + (size_t)bc0 * DIN)
                                              : (wv + (size_t)(bc0 - DKV) * DIN);
    const unsigned short* wsrc1 = (bc1 < DKV) ? (wk + (size_t)bc1 * DIN)
                                              : (wv + (size_t)(bc1 - DKV) * DIN);

    v8f zero8 = {0.f, 0.f, 0.f, 0.f, 0.f, 0.f, 0.f, 0.f};
    v8f acc[2][2];
    acc[0][0] = zero8; acc[0][1] = zero8; acc[1][0] = zero8; acc[1][1] = zero8;

    int wr = w >> 2, wc = w & 3;       // wave position: 2 x 4
    int lr = l & 15, h = l >> 4;
    int koA = h * 8, koB = h * 8;

    for (int k0 = 0; k0 < DIN; k0 += 32) {
        __syncthreads();
        {   // stage A with fused LayerNorm -> bf16
            float4 f4 = *reinterpret_cast<const float4*>(feat + (R0 + ar0) * DIN + k0 + ak0);
            float xs[4] = {f4.x, f4.y, f4.z, f4.w};
#pragma unroll
            for (int j = 0; j < 4; j++) {
                int kk = k0 + ak0 + j;
                As[ar0 * 32 + ak0 + j] = f2bf((xs[j] - mu0) * rs0 * g[kk] + bia[kk]);
            }
            f4 = *reinterpret_cast<const float4*>(feat + (R0 + ar1) * DIN + k0 + ak0);
            float ys[4] = {f4.x, f4.y, f4.z, f4.w};
#pragma unroll
            for (int j = 0; j < 4; j++) {
                int kk = k0 + ak0 + j;
                As[ar1 * 32 + ak0 + j] = f2bf((ys[j] - mu1) * rs1 * g[kk] + bia[kk]);
            }
        }
        // stage B (bf16 weight rows)
        *reinterpret_cast<uint4*>(&Bs[br0 * 32 + bk0]) =
            *reinterpret_cast<const uint4*>(wsrc0 + k0 + bk0);
        *reinterpret_cast<uint4*>(&Bs[br1 * 32 + bk0]) =
            *reinterpret_cast<const uint4*>(wsrc1 + k0 + bk0);
        if (k0 + 32 < DIN) {
            __builtin_prefetch(feat + (R0 + ar0) * DIN + k0 + 32 + ak0, 0, 0);
            __builtin_prefetch(feat + (R0 + ar1) * DIN + k0 + 32 + ak0, 0, 0);
            __builtin_prefetch(wsrc0 + k0 + 32 + bk0, 0, 0);
            __builtin_prefetch(wsrc1 + k0 + 32 + bk0, 0, 0);
        }
        __syncthreads();

        FragAB a0, a1, b0, b1;
        const uint4* p;
        p = reinterpret_cast<const uint4*>(&As[(wr * 32 + lr) * 32 + koA]);
        a0.q[0] = p[0]; a0.q[1] = p[2];
        p = reinterpret_cast<const uint4*>(&As[(wr * 32 + 16 + lr) * 32 + koA]);
        a1.q[0] = p[0]; a1.q[1] = p[2];
        p = reinterpret_cast<const uint4*>(&Bs[(wc * 32 + lr) * 32 + koB]);
        b0.q[0] = p[0]; b0.q[1] = p[2];
        p = reinterpret_cast<const uint4*>(&Bs[(wc * 32 + 16 + lr) * 32 + koB]);
        b1.q[0] = p[0]; b1.q[1] = p[2];

        acc[0][0] = __builtin_amdgcn_wmma_f32_16x16x32_bf16(false, a0.v, false, b0.v, (short)0, acc[0][0], false, false);
        acc[0][1] = __builtin_amdgcn_wmma_f32_16x16x32_bf16(false, a0.v, false, b1.v, (short)0, acc[0][1], false, false);
        acc[1][0] = __builtin_amdgcn_wmma_f32_16x16x32_bf16(false, a1.v, false, b0.v, (short)0, acc[1][0], false, false);
        acc[1][1] = __builtin_amdgcn_wmma_f32_16x16x32_bf16(false, a1.v, false, b1.v, (short)0, acc[1][1], false, false);
    }

#pragma unroll
    for (int mi = 0; mi < 2; mi++) {
#pragma unroll
        for (int ni = 0; ni < 2; ni++) {
#pragma unroll
            for (int r = 0; r < 8; r++) {
                int m = wr * 32 + mi * 16 + r + 8 * h;
                int n = C0 + wc * 32 + ni * 16 + lr;
                long long gr = R0 + m;
                unsigned short hv = f2bf(acc[mi][ni][r]);
                if (n < DKV) {
                    kbf[(size_t)gr * DKV + n] = hv;
                } else {
                    int bb = (int)(gr >> 12), ff = (int)(gr & (FN - 1));
                    vtbf[(size_t)bb * DKV * FN + (size_t)(n - DKV) * FN + ff] = hv;
                }
            }
        }
    }
}

// ---------------- slot LayerNorm + q projection (+ zero attn sums) ---------
__global__ void k_slotq(const float* __restrict__ slots, const float* __restrict__ g,
                        const float* __restrict__ b, const float* __restrict__ wq,
                        float* __restrict__ sn, unsigned short* __restrict__ qbf,
                        float* __restrict__ asum) {
    __shared__ float sL[SN * DSL];
    int t = threadIdx.x, w = t >> 5, l = t & 31;
    int bidx = blockIdx.x;
    const float* sb = slots + (size_t)bidx * SN * DSL;
    for (int row = w * 2; row < w * 2 + 2; ++row) {
        const float* xr = sb + row * DSL;
        float s = 0.f, sq = 0.f;
        for (int i = l; i < DSL; i += 32) { float v = xr[i]; s += v; sq += v * v; }
        s = wred(s); sq = wred(sq);
        float mu = s / (float)DSL, var = sq / (float)DSL - mu * mu;
        float rs = rsqrtf(var + LN_EPS_F);
        for (int i = l; i < DSL; i += 32) {
            float v = (xr[i] - mu) * rs * g[i] + b[i];
            sL[row * DSL + i] = v;
            sn[((size_t)bidx * SN + row) * DSL + i] = v;
        }
    }
    if (t < SN) asum[bidx * SN + t] = 0.f;
    __syncthreads();
    for (int i = 0; i < 32; i++) {
        int idx = t + 256 * i, s = idx >> 9, n = idx & (DSL - 1);
        const float* wr = wq + (size_t)n * DSL;
        float a = 0.f;
        for (int k = 0; k < DSL; k++) a += sL[s * DSL + k] * wr[k];
        qbf[((size_t)bidx * SN + s) * DKV + n] = f2bf(a);
    }
}

// ---------------- dots (WMMA, A-frag reused over 4 N-tiles) + softmax ------
__global__ void k_dots(const unsigned short* __restrict__ qbf,
                       const unsigned short* __restrict__ kbf,
                       unsigned short* __restrict__ attnbf,
                       float* __restrict__ attn_out, float* __restrict__ asum) {
    int t = threadIdx.x, w = t >> 5, l = t & 31;
    int bidx = blockIdx.x;
    int fbase = blockIdx.y * 256 + w * 64;   // 4 waves x 4 tiles = 256 features/block
    const unsigned short* qb = qbf + (size_t)bidx * SN * DKV;
    const unsigned short* kb = kbf + (size_t)bidx * FN * DKV;
    v8f zero8 = {0.f, 0.f, 0.f, 0.f, 0.f, 0.f, 0.f, 0.f};
    v8f acc[4];
    acc[0] = zero8; acc[1] = zero8; acc[2] = zero8; acc[3] = zero8;
    int row = l & 15, koA = (l >> 4) * 8, koB = (l >> 4) * 8;

    for (int k0 = 0; k0 < DKV; k0 += 32) {
        FragAB a;
        const uint4* pa = (const uint4*)(qb + (size_t)row * DKV + k0 + koA);
        a.q[0] = pa[0]; a.q[1] = pa[2];
#pragma unroll
        for (int ni = 0; ni < 4; ni++) {
            FragAB b;
            const uint4* pb = (const uint4*)(kb + (size_t)(fbase + ni * 16 + row) * DKV + k0 + koB);
            b.q[0] = pb[0]; b.q[1] = pb[2];
            if (k0 + 32 < DKV)
                __builtin_prefetch(kb + (size_t)(fbase + ni * 16 + row) * DKV + k0 + 32 + koB, 0, 0);
            acc[ni] = __builtin_amdgcn_wmma_f32_16x16x32_bf16(false, a.v, false, b.v,
                                                              (short)0, acc[ni], false, false);
        }
    }

    const float scale = 0.044194173824159216f; // 512**-0.5
    int h = l >> 4;
#pragma unroll
    for (int ni = 0; ni < 4; ni++) {
        int f0 = fbase + ni * 16;
        float x[8], mx = -1e30f;
#pragma unroll
        for (int r = 0; r < 8; r++) { x[r] = acc[ni][r] * scale; mx = fmaxf(mx, x[r]); }
        mx = fmaxf(mx, __shfl_xor(mx, 16, 32));   // column split across lanes (l, l^16)
        float se = 0.f;
#pragma unroll
        for (int r = 0; r < 8; r++) { x[r] = __expf(x[r] - mx); se += x[r]; }
        se += __shfl_xor(se, 16, 32);
        float inv = 1.f / se;
#pragma unroll
        for (int r = 0; r < 8; r++) {
            int m = r + 8 * h;
            float p = x[r] * inv;
            size_t oi = ((size_t)bidx * SN + m) * FN + f0 + row;
            attn_out[oi] = p;               // pre_norm_attn output
            float pe = p + EPS_A;
            attnbf[oi] = f2bf(pe);          // A-matrix for updates GEMM
            float v = pe;                   // sum over features within 16-lane group
            v += __shfl_xor(v, 1, 32); v += __shfl_xor(v, 2, 32);
            v += __shfl_xor(v, 4, 32); v += __shfl_xor(v, 8, 32);
            if ((l & 15) == 0) atomicAdd(&asum[bidx * SN + m], v);
        }
    }
}

// ---------------- updates = attn @ values (WMMA, A reused x4), fold 1/sum --
__global__ void k_upd(const unsigned short* __restrict__ attnbf,
                      const unsigned short* __restrict__ vtbf,
                      const float* __restrict__ asum, float* __restrict__ upd) {
    int t = threadIdx.x, w = t >> 5, l = t & 31;
    int bidx = blockIdx.x;
    int dbase = blockIdx.y * 256 + w * 64;   // 4 waves x 4 tiles = 256 dims/block
    const unsigned short* ab = attnbf + (size_t)bidx * SN * FN;
    const unsigned short* vb = vtbf + (size_t)bidx * DKV * FN;
    v8f zero8 = {0.f, 0.f, 0.f, 0.f, 0.f, 0.f, 0.f, 0.f};
    v8f acc[4];
    acc[0] = zero8; acc[1] = zero8; acc[2] = zero8; acc[3] = zero8;
    int row = l & 15, koA = (l >> 4) * 8, koB = (l >> 4) * 8;

    for (int k0 = 0; k0 < FN; k0 += 32) {
        FragAB a;
        const uint4* pa = (const uint4*)(ab + (size_t)row * FN + k0 + koA);
        a.q[0] = pa[0]; a.q[1] = pa[2];
#pragma unroll
        for (int ni = 0; ni < 4; ni++) {
            FragAB b;
            const uint4* pb = (const uint4*)(vb + (size_t)(dbase + ni * 16 + row) * FN + k0 + koB);
            b.q[0] = pb[0]; b.q[1] = pb[2];
            if (k0 + 32 < FN)
                __builtin_prefetch(vb + (size_t)(dbase + ni * 16 + row) * FN + k0 + 32 + koB, 0, 0);
            acc[ni] = __builtin_amdgcn_wmma_f32_16x16x32_bf16(false, a.v, false, b.v,
                                                              (short)0, acc[ni], false, false);
        }
    }
    int h = l >> 4;
#pragma unroll
    for (int ni = 0; ni < 4; ni++) {
#pragma unroll
        for (int r = 0; r < 8; r++) {
            int m = r + 8 * h;
            float s = asum[bidx * SN + m];
            upd[((size_t)bidx * SN + m) * DSL + dbase + ni * 16 + row] = acc[ni][r] / s;
        }
    }
}

// ---------------- GRU cell: slots = (1-z)*n + z*s_n ------------------------
__global__ void k_gru(const float* __restrict__ upd, const float* __restrict__ sn,
                      const float* __restrict__ wih, const float* __restrict__ whh,
                      const float* __restrict__ bih, const float* __restrict__ bhh,
                      float* __restrict__ slots_ws) {
    __shared__ float uL[SN * DSL];
    __shared__ float hL[SN * DSL];
    int t = threadIdx.x, bidx = blockIdx.x;
    for (int i = t; i < SN * DSL; i += 256) {
        uL[i] = upd[(size_t)bidx * SN * DSL + i];
        hL[i] = sn[(size_t)bidx * SN * DSL + i];
    }
    __syncthreads();
    for (int i = 0; i < 32; i++) {
        int idx = t + 256 * i, s = idx >> 9, d = idx & (DSL - 1);
        const float* u = &uL[s * DSL];
        const float* hh = &hL[s * DSL];
        float gr = bih[d], gz = bih[DSL + d], gn = bih[2 * DSL + d];
        float hr = bhh[d], hz = bhh[DSL + d], hn = bhh[2 * DSL + d];
        const float* wr_ = wih + (size_t)d * DSL;
        const float* wz_ = wih + (size_t)(DSL + d) * DSL;
        const float* wn_ = wih + (size_t)(2 * DSL + d) * DSL;
        const float* vr_ = whh + (size_t)d * DSL;
        const float* vz_ = whh + (size_t)(DSL + d) * DSL;
        const float* vn_ = whh + (size_t)(2 * DSL + d) * DSL;
        for (int k = 0; k < DSL; k++) {
            float uk = u[k], hk = hh[k];
            gr += uk * wr_[k]; gz += uk * wz_[k]; gn += uk * wn_[k];
            hr += hk * vr_[k]; hz += hk * vz_[k]; hn += hk * vn_[k];
        }
        float r = 1.f / (1.f + __expf(-(gr + hr)));
        float z = 1.f / (1.f + __expf(-(gz + hz)));
        float nn = tanhf(gn + r * hn);
        slots_ws[(size_t)bidx * SN * DSL + idx] = (1.f - z) * nn + z * hh[d];
    }
}

// ---------------- residual MLP with LayerNorm ------------------------------
__global__ void k_mlp(float* __restrict__ slots_io, const float* __restrict__ g,
                      const float* __restrict__ b, const float* __restrict__ w1,
                      const float* __restrict__ b1, const float* __restrict__ w2,
                      const float* __restrict__ b2, float* __restrict__ out_slots) {
    __shared__ float tL[SN * DSL];
    __shared__ float hidL[SN * 256];
    int t = threadIdx.x, w = t >> 5, l = t & 31;
    int bidx = blockIdx.x;
    const float* sb = slots_io + (size_t)bidx * SN * DSL;
    for (int row = w * 2; row < w * 2 + 2; ++row) {
        const float* xr = sb + row * DSL;
        float s = 0.f, sq = 0.f;
        for (int i = l; i < DSL; i += 32) { float v = xr[i]; s += v; sq += v * v; }
        s = wred(s); sq = wred(sq);
        float mu = s / (float)DSL, var = sq / (float)DSL - mu * mu;
        float rs = rsqrtf(var + LN_EPS_F);
        for (int i = l; i < DSL; i += 32)
            tL[row * DSL + i] = (xr[i] - mu) * rs * g[i] + b[i];
    }
    __syncthreads();
    float acc[32];
#pragma unroll
    for (int i = 0; i < 32; i++) acc[i] = 0.f;
    for (int c = 0; c < 8; c++) {
        int jb = c * 256;
        for (int i = 0; i < 16; i++) {
            int idx = t + 256 * i, s = idx >> 8, jj = idx & 255;
            const float* wr_ = w1 + (size_t)(jb + jj) * DSL;
            float a = b1[jb + jj];
            for (int k = 0; k < DSL; k++) a += tL[s * DSL + k] * wr_[k];
            hidL[s * 256 + jj] = fmaxf(a, 0.f);
        }
        __syncthreads();
        for (int i = 0; i < 32; i++) {
            int idx = t + 256 * i, s = idx >> 9, d = idx & (DSL - 1);
            const float* w2r = w2 + (size_t)d * DHID + jb;
            const float* hr = &hidL[s * 256];
            float a = 0.f;
            for (int k = 0; k < 256; k++) a += hr[k] * w2r[k];
            acc[i] += a;
        }
        __syncthreads();
    }
    for (int i = 0; i < 32; i++) {
        int idx = t + 256 * i, d = idx & (DSL - 1);
        float v = sb[idx] + acc[i] + b2[d];
        slots_io[(size_t)bidx * SN * DSL + idx] = v;
        out_slots[(size_t)bidx * SN * DSL + idx] = v;
    }
}

extern "C" void kernel_launch(void* const* d_in, const int* in_sizes, int n_in,
                              void* d_out, int out_size, void* d_ws, size_t ws_size,
                              hipStream_t stream) {
    const float* slots0 = (const float*)d_in[0];
    const float* feat   = (const float*)d_in[1];
    const float* lnfg   = (const float*)d_in[2];
    const float* lnfb   = (const float*)d_in[3];
    const float* lnsg   = (const float*)d_in[4];
    const float* lnsb   = (const float*)d_in[5];
    const float* w_k    = (const float*)d_in[6];
    const float* w_v    = (const float*)d_in[7];
    const float* w_q    = (const float*)d_in[8];
    const float* wih    = (const float*)d_in[9];
    const float* whh    = (const float*)d_in[10];
    const float* bih    = (const float*)d_in[11];
    const float* bhh    = (const float*)d_in[12];
    const float* mlng   = (const float*)d_in[13];
    const float* mlnb   = (const float*)d_in[14];
    const float* w1     = (const float*)d_in[15];
    const float* b1     = (const float*)d_in[16];
    const float* w2     = (const float*)d_in[17];
    const float* b2     = (const float*)d_in[18];
    float* out_slots = (float*)d_out;
    float* out_attn  = (float*)d_out + (size_t)BN * SN * DSL;

    char* ws = (char*)d_ws;
    size_t off = 0;
    auto walloc = [&](size_t bytes) -> void* {
        void* p = ws + off;
        off += (bytes + 255) & ~(size_t)255;
        return p;
    };
    float*          stats  = (float*)walloc((size_t)BN * FN * 2 * sizeof(float));
    unsigned short* wkbf   = (unsigned short*)walloc((size_t)DKV * DIN * 2);
    unsigned short* wvbf   = (unsigned short*)walloc((size_t)DKV * DIN * 2);
    unsigned short* qbf    = (unsigned short*)walloc((size_t)BN * SN * DKV * 2);
    float*          snbuf  = (float*)walloc((size_t)BN * SN * DSL * sizeof(float));
    unsigned short* attnbf = (unsigned short*)walloc((size_t)BN * SN * FN * 2);
    float*          asum   = (float*)walloc((size_t)BN * SN * sizeof(float));
    float*          updbuf = (float*)walloc((size_t)BN * SN * DSL * sizeof(float));
    float*          slotsW = (float*)walloc((size_t)BN * SN * DSL * sizeof(float));
    unsigned short* kbf    = (unsigned short*)walloc((size_t)BN * FN * DKV * 2);
    unsigned short* vtbf   = (unsigned short*)walloc((size_t)BN * DKV * FN * 2);
    (void)ws_size; (void)in_sizes; (void)n_in; (void)out_size;

    k_feat_stats<<<(BN * FN) / 8, 256, 0, stream>>>(feat, stats);
    k_cvt2<<<(DKV * DIN + 255) / 256, 256, 0, stream>>>(w_k, w_v, wkbf, wvbf);
    k_kv<<<dim3((BN * FN) / 64, (2 * DKV) / 128), 256, 0, stream>>>(
        feat, stats, lnfg, lnfb, wkbf, wvbf, kbf, vtbf);

    for (int it = 0; it < 3; ++it) {
        const float* ssrc = (it == 0) ? slots0 : slotsW;
        k_slotq<<<BN, 256, 0, stream>>>(ssrc, lnsg, lnsb, w_q, snbuf, qbf, asum);
        k_dots<<<dim3(BN, FN / 256), 128, 0, stream>>>(qbf, kbf, attnbf, out_attn, asum);
        k_upd<<<dim3(BN, DSL / 256), 128, 0, stream>>>(attnbf, vtbf, asum, updbuf);
        k_gru<<<BN, 256, 0, stream>>>(updbuf, snbuf, wih, whh, bih, bhh, slotsW);
        k_mlp<<<BN, 256, 0, stream>>>(slotsW, mlng, mlnb, w1, b1, w2, b2, out_slots);
    }
}
